// SphericalHarmonicsIPA_84464826843859
// MI455X (gfx1250) — matrix operand order repaired and checked by reading, not compiled, not run on hardware
//
#include <hip/hip_runtime.h>

// ---------------------------------------------------------------------------
// SphericalHarmonicsIPA on MI455X (gfx1250, wave32, WMMA bf16/f32-acc)
// B=2 N=1024 CS=512 CZ=64 CH=128 H=8 L=3 RB=8 -> SH=224, DZ=16
// ---------------------------------------------------------------------------

typedef __attribute__((ext_vector_type(16))) __bf16 v16bf;
typedef __attribute__((ext_vector_type(8)))  float  v8f;

#define B_   2
#define N_   1024
#define CS_  512
#define CZ_  64
#define CH_  128
#define H_   8
#define SH_  224
#define DZ_  16
#define QKW_ 352            // CH + SH
#define VW_  368            // CH + SH + DZ
#define OCAT_ 2944          // H * VW
#define INVSQ_CH_ 0.08838834764831845f   // 1/sqrt(128)
#define INVSQ_SH_ 0.06681531047810609f   // 1/sqrt(224)

__device__ __forceinline__ v8f wmma_bf16(v16bf a, v16bf b, v8f c) {
  return __builtin_amdgcn_wmma_f32_16x16x32_bf16(false, a, false, b,
                                                 (short)0, c, false, false);
}

// ---- fragment loaders (contiguous K = 16g+e in both A and B) ----------------------
// A: 16x32 (MxK), row-major source. lanes 0-15: M=lane, K 0..15;
// lanes 16-31 hold K 16..31 (element e -> K=16g+e).
__device__ __forceinline__ v16bf load_a_f32(const float* A, int lda,
                                            int row0, int k0, int lane) {
  int g = (lane >> 4) & 1, m = lane & 15;
  const float* p = A + (size_t)(row0 + m) * lda + k0 + 16 * g;
  v16bf a;
#pragma unroll
  for (int e = 0; e < 16; ++e) {
    int kl = e;
    a[e] = (__bf16)p[kl];
  }
  return a;
}
__device__ __forceinline__ v16bf load_a_bf16(const __bf16* A, int lda,
                                             int row0, int k0, int lane) {
  int g = (lane >> 4) & 1, m = lane & 15;
  const __bf16* p = A + (size_t)(row0 + m) * lda + k0 + 16 * g;
  v16bf a;
#pragma unroll
  for (int e = 0; e < 16; ++e) {
    int kl = e;
    a[e] = p[kl];
  }
  return a;
}
// B: 32x16 (KxN) from row-major [K x N]: VGPR v holds K=16g+2v,+1 for col lane&15
__device__ __forceinline__ v16bf load_b_f32(const float* Bm, int ldb,
                                            int k0, int col0, int lane) {
  int g = (lane >> 4) & 1, n = lane & 15;
  const float* p = Bm + (size_t)(k0 + 16 * g) * ldb + col0 + n;
  v16bf b;
#pragma unroll
  for (int e = 0; e < 16; ++e) b[e] = (__bf16)p[(size_t)e * ldb];
  return b;
}
__device__ __forceinline__ v16bf load_b_bf16(const __bf16* Bm, int ldb,
                                             int k0, int col0, int lane) {
  int g = (lane >> 4) & 1, n = lane & 15;
  const __bf16* p = Bm + (size_t)(k0 + 16 * g) * ldb + col0 + n;
  v16bf b;
#pragma unroll
  for (int e = 0; e < 16; ++e) b[e] = p[(size_t)e * ldb];
  return b;
}
// B = X^T where X is row-major [Ncols x K] (e.g. K-matrix for Q*K^T):
// lane (g,n) reads 16 consecutive K elements of row col0+n.
__device__ __forceinline__ v16bf load_bT_bf16(const __bf16* X, int ldx,
                                              int k0, int col0, int lane) {
  int g = (lane >> 4) & 1, n = lane & 15;
  const __bf16* p = X + (size_t)(col0 + n) * ldx + k0 + 16 * g;
  v16bf b;
#pragma unroll
  for (int e = 0; e < 16; ++e) b[e] = p[e];
  return b;
}

// ---------------------------------------------------------------------------
// Generic GEMM + bias with 4x M-tile register reuse:
// each wave computes a 64-row x 16-col block; B fragment loaded once per
// k-step and fed to 4 WMMAs. Requires M % 64 == 0 (M = 2048 here).
// ---------------------------------------------------------------------------
__global__ void gemm_bias_kernel(const float* __restrict__ A,
                                 const float* __restrict__ W,
                                 const float* __restrict__ bias,
                                 float* __restrict__ D,
                                 int M, int N, int K) {
  int lane = threadIdx.x & 31;
  int wave = threadIdx.x >> 5;
  int tiles_n = N >> 4;
  int mgroups = M >> 6;
  int tile = blockIdx.x * (blockDim.x >> 5) + wave;
  if (tile >= mgroups * tiles_n) return;
  int row0 = (tile / tiles_n) << 6;
  int col0 = (tile % tiles_n) << 4;
  v8f acc[4] = {v8f{}, v8f{}, v8f{}, v8f{}};
  for (int k0 = 0; k0 < K; k0 += 32) {
    v16bf b = load_b_f32(W, N, k0, col0, lane);
#pragma unroll
    for (int mt = 0; mt < 4; ++mt) {
      v16bf a = load_a_f32(A, K, row0 + 16 * mt, k0, lane);
      acc[mt] = wmma_bf16(a, b, acc[mt]);
    }
  }
  int g = lane >> 4, n = lane & 15;
  float bv = bias[col0 + n];
#pragma unroll
  for (int mt = 0; mt < 4; ++mt)
#pragma unroll
    for (int r = 0; r < 8; ++r)
      D[(size_t)(row0 + 16 * mt + r + 8 * g) * N + col0 + n] = acc[mt][r] + bv;
}

// ---------------------------------------------------------------------------
// z fused: single streaming pass over z (512 MB = the HBM floor of this op).
// Double-buffered CDNA5 async staging: GLOBAL_LOAD_ASYNC_TO_LDS_B128 streams
// chunk c+1 into one LDS buffer while VALU consumes chunk c from the other.
// ASYNCcnt completes in order, so "s_wait_asynccnt 16" guarantees the current
// buffer's 16 loads landed while the next buffer's 16 remain in flight.
// Rows padded to 68 dwords to spread the 64 LDS banks.
//   bias2[b,h,i,j] = z[b,i,j,:]@Wb[:,h] + bb[h] + INF*(mask_i*mask_j - 1)
//   pz[b,i,d]      = sum_j z[b,i,j,:]@Wdz[:,d] + N*bdz[d]
// ---------------------------------------------------------------------------
#define ZROW_PAD_ 68    // dwords per staged row (64 + 4 pad)
#define ZCHUNK_   256   // j-rows per chunk (4 chunks of N=1024)

__device__ __forceinline__ void z_async_issue(unsigned lds_base, unsigned long long src,
                                              int t) {
#pragma unroll
  for (int it = 0; it < 16; ++it) {
    int qq = t + 256 * it;                  // dword4 index within chunk
    int jl = qq >> 4, c4 = qq & 15;
    unsigned lds_off = lds_base + (unsigned)((jl * ZROW_PAD_ + c4 * 4) * 4);
    unsigned long long ga = src + (unsigned long long)qq * 16ull;
    asm volatile("global_load_async_to_lds_b128 %0, %1, off"
                 :: "v"(lds_off), "v"(ga) : "memory");
  }
}

__global__ void z_fused_kernel(const float* __restrict__ z,
                               const float* __restrict__ mask,
                               const float* __restrict__ Wb,
                               const float* __restrict__ bb,
                               const float* __restrict__ Wdz,
                               const float* __restrict__ bdz,
                               float* __restrict__ bias2,
                               float* __restrict__ pz) {
  extern __shared__ float zbuf[];          // 2 * 256 * 68 dwords = 139264 bytes
  __shared__ float sWb[CZ_ * H_];
  __shared__ float sWdz[CZ_ * DZ_];
  __shared__ float sbb[H_];
  __shared__ float spz[DZ_];
  int b = blockIdx.x >> 10;
  int i = blockIdx.x & (N_ - 1);
  int t = threadIdx.x;
  for (int x = t; x < CZ_ * H_; x += 256) sWb[x] = Wb[x];
  for (int x = t; x < CZ_ * DZ_; x += 256) sWdz[x] = Wdz[x];
  if (t < H_) sbb[t] = bb[t];
  if (t < DZ_) spz[t] = 0.f;

  float maski = mask[b * N_ + i];
  float pzl[DZ_];
#pragma unroll
  for (int d = 0; d < DZ_; ++d) pzl[d] = 0.f;

  unsigned zbase = (unsigned)(unsigned long long)(void*)zbuf;  // LDS byte offset
  const unsigned bufbytes = ZCHUNK_ * ZROW_PAD_ * 4;
  const float* zrow = z + ((size_t)b * N_ + i) * (size_t)N_ * CZ_;
  unsigned long long zr64 = (unsigned long long)(const void*)zrow;

  // prologue: stage chunk 0 into buffer 0
  z_async_issue(zbase, zr64, t);

  const int NCH = N_ / ZCHUNK_;  // 4
  for (int c = 0; c < NCH; ++c) {
    int cur = c & 1;
    if (c + 1 < NCH) {
      // stream next chunk into the other buffer, then wait only for current
      z_async_issue(zbase + ((c + 1) & 1) * bufbytes,
                    zr64 + (unsigned long long)(c + 1) * ZCHUNK_ * CZ_ * 4ull, t);
      asm volatile("s_wait_asynccnt 0x10" ::: "memory");  // first 16 landed
    } else {
      asm volatile("s_wait_asynccnt 0x0" ::: "memory");
    }
    __syncthreads();

    // ---- compute: thread t handles j = c*256 + t from LDS ----
    int j = c * ZCHUNK_ + t;
    const float4* zp4 = (const float4*)(zbuf + cur * (ZCHUNK_ * ZROW_PAD_) + t * ZROW_PAD_);
    float acc_b[H_], acc_d[DZ_];
#pragma unroll
    for (int h = 0; h < H_; ++h) acc_b[h] = 0.f;
#pragma unroll
    for (int d = 0; d < DZ_; ++d) acc_d[d] = 0.f;
#pragma unroll
    for (int c4 = 0; c4 < CZ_ / 4; ++c4) {
      float4 zv = zp4[c4];
      float comp[4] = {zv.x, zv.y, zv.z, zv.w};
#pragma unroll
      for (int u = 0; u < 4; ++u) {
        int cc = c4 * 4 + u;
        float v = comp[u];
#pragma unroll
        for (int h = 0; h < H_; ++h) acc_b[h] += v * sWb[cc * H_ + h];
#pragma unroll
        for (int d = 0; d < DZ_; ++d) acc_d[d] += v * sWdz[cc * DZ_ + d];
      }
    }
    float sq = 100000.f * (maski * mask[b * N_ + j] - 1.f);
#pragma unroll
    for (int h = 0; h < H_; ++h)
      bias2[(((size_t)(b * H_ + h) * N_ + i) * N_) + j] = acc_b[h] + sbb[h] + sq;
#pragma unroll
    for (int d = 0; d < DZ_; ++d) pzl[d] += acc_d[d];
    __syncthreads();   // all waves done reading `cur` before it is re-issued
  }
#pragma unroll
  for (int d = 0; d < DZ_; ++d) atomicAdd(&spz[d], pzl[d]);
  __syncthreads();
  if (t < DZ_)
    pz[((size_t)b * N_ + i) * DZ_ + t] = spz[t] + (float)N_ * bdz[t];
}

// ---------------------------------------------------------------------------
// Wigner-rotated SH element. Per (b,n,h) 224-vector, layout d = l*56+m*8+rb:
//  - m >= 2l+1           -> 0   (reference zero-fills)
//  - l==1 (d in [56,80)) -> reference's raw reshape == rotate each consecutive
//                           triple of the 24 elements by r (3x3)
//  - else copy.
// ---------------------------------------------------------------------------
__device__ __forceinline__ float sh_elem(const float* __restrict__ shp,
                                         const float* __restrict__ r9, int d) {
  int l = d / 56;
  int m = (d % 56) >> 3;
  if (m >= 2 * l + 1) return 0.f;
  if (l == 1) {
    int pos = d - 56;
    int gt = pos / 3;
    int ii = pos - gt * 3;
    const float* xp = shp + 56 + gt * 3;
    const float* rr = r9 + ii * 3;
    return rr[0] * xp[0] + rr[1] * xp[1] + rr[2] * xp[2];
  }
  return shp[d];
}

// Qcat[b,h,n,0:352] = [ q*1/sqrt(CH) | wigner(q_sh)*1/sqrt(SH) ]  (bf16)
__global__ void pack_q_kernel(const float* __restrict__ q,
                              const float* __restrict__ qsh,
                              const float* __restrict__ r,
                              __bf16* __restrict__ Qcat) {
  long idx = (long)blockIdx.x * blockDim.x + threadIdx.x;
  if (idx >= (long)B_ * H_ * N_ * QKW_) return;
  int c = idx % QKW_;
  int n = (idx / QKW_) % N_;
  int h = (idx / ((long)QKW_ * N_)) % H_;
  int b = idx / ((long)QKW_ * N_ * H_);
  float val;
  if (c < CH_) {
    val = q[(((size_t)(b * N_ + n) * H_ + h) << 7) + c] * INVSQ_CH_;
  } else {
    const float* shp = qsh + ((size_t)(b * N_ + n) * H_ + h) * SH_;
    const float* r9 = r + ((size_t)b * N_ + n) * 9;
    val = sh_elem(shp, r9, c - CH_) * INVSQ_SH_;
  }
  Qcat[idx] = (__bf16)val;
}

// Kcat[b,h,n,0:352] = [ k | wigner(k_sh) ]   (k = kv[...,0:128], first-H sh)
__global__ void pack_k_kernel(const float* __restrict__ kv,
                              const float* __restrict__ kvsh,
                              const float* __restrict__ r,
                              __bf16* __restrict__ Kcat) {
  long idx = (long)blockIdx.x * blockDim.x + threadIdx.x;
  if (idx >= (long)B_ * H_ * N_ * QKW_) return;
  int c = idx % QKW_;
  int n = (idx / QKW_) % N_;
  int h = (idx / ((long)QKW_ * N_)) % H_;
  int b = idx / ((long)QKW_ * N_ * H_);
  float val;
  if (c < CH_) {
    val = kv[((size_t)(b * N_ + n) * H_ + h) * 256 + c];
  } else {
    const float* shp = kvsh + ((size_t)(b * N_ + n) * 16 + h) * SH_;
    const float* r9 = r + ((size_t)b * N_ + n) * 9;
    val = sh_elem(shp, r9, c - CH_);
  }
  Kcat[idx] = (__bf16)val;
}

// Vcat[b,h,n,0:368] = [ v | wigner(v_sh) | pz ]  (v = kv[...,128:], second-H sh)
__global__ void pack_v_kernel(const float* __restrict__ kv,
                              const float* __restrict__ kvsh,
                              const float* __restrict__ r,
                              const float* __restrict__ pz,
                              __bf16* __restrict__ Vcat) {
  long idx = (long)blockIdx.x * blockDim.x + threadIdx.x;
  if (idx >= (long)B_ * H_ * N_ * VW_) return;
  int c = idx % VW_;
  int n = (idx / VW_) % N_;
  int h = (idx / ((long)VW_ * N_)) % H_;
  int b = idx / ((long)VW_ * N_ * H_);
  float val;
  if (c < CH_) {
    val = kv[((size_t)(b * N_ + n) * H_ + h) * 256 + CH_ + c];
  } else if (c < CH_ + SH_) {
    const float* shp = kvsh + ((size_t)(b * N_ + n) * 16 + H_ + h) * SH_;
    const float* r9 = r + ((size_t)b * N_ + n) * 9;
    val = sh_elem(shp, r9, c - CH_);
  } else {
    val = pz[((size_t)b * N_ + n) * DZ_ + (c - CH_ - SH_)];
  }
  Vcat[idx] = (__bf16)val;
}

// ---------------------------------------------------------------------------
// Attention scores + softmax. Block = (b,h, 16-row i-tile), 8 waves.
// Waves cooperatively fill 16x1024 f32 scores in LDS (64 KB of 320 KB WGP LDS),
// S = Qcat @ Kcat^T (11 bf16 WMMA per 16x16 tile; Q fragments register-resident
// for the whole row) + bias2, then 16-lane-shuffle softmax, bf16 prob write.
// ---------------------------------------------------------------------------
__global__ void attn_scores_kernel(const __bf16* __restrict__ Qcat,
                                   const __bf16* __restrict__ Kcat,
                                   const float* __restrict__ bias2,
                                   __bf16* __restrict__ Aout) {
  extern __shared__ float sc[];  // [16][1024]
  int lane = threadIdx.x & 31;
  int wave = threadIdx.x >> 5;
  int bid = blockIdx.x;
  int it = bid & 63;
  int h = (bid >> 6) & 7;
  int b = bid >> 9;
  int i0 = it << 4;

  const __bf16* Qb = Qcat + (size_t)(b * H_ + h) * N_ * QKW_;
  const __bf16* Kb = Kcat + (size_t)(b * H_ + h) * N_ * QKW_;
  const float* biasrow = bias2 + ((size_t)(b * H_ + h) * N_ + i0) * N_;

  v16bf afr[11];
#pragma unroll
  for (int kk = 0; kk < 11; ++kk)
    afr[kk] = load_a_bf16(Qb, QKW_, i0, kk * 32, lane);

  for (int jt = wave; jt < 64; jt += 8) {
    int j0 = jt << 4;
    v8f acc = {};
#pragma unroll
    for (int kk = 0; kk < 11; ++kk) {
      v16bf bfr = load_bT_bf16(Kb, QKW_, kk * 32, j0, lane);
      acc = wmma_bf16(afr[kk], bfr, acc);
    }
    int g = lane >> 4, n4 = lane & 15;
#pragma unroll
    for (int rr = 0; rr < 8; ++rr) {
      int mrow = rr + 8 * g;
      sc[mrow * N_ + j0 + n4] = acc[rr] + biasrow[(size_t)mrow * N_ + j0 + n4];
    }
  }
  __syncthreads();

  int row = threadIdx.x >> 4;   // 16 threads per row, rows within one wave
  int t16 = threadIdx.x & 15;
  float mx = -3.0e38f;
  for (int j = t16; j < N_; j += 16) mx = fmaxf(mx, sc[row * N_ + j]);
#pragma unroll
  for (int o = 8; o > 0; o >>= 1) mx = fmaxf(mx, __shfl_xor(mx, o, 16));
  float sum = 0.f;
  for (int j = t16; j < N_; j += 16) {
    float e = __expf(sc[row * N_ + j] - mx);
    sc[row * N_ + j] = e;
    sum += e;
  }
#pragma unroll
  for (int o = 8; o > 0; o >>= 1) sum += __shfl_xor(sum, o, 16);
  float inv = 1.0f / sum;
  __bf16* arow = Aout + ((size_t)(b * H_ + h) * N_ + i0 + row) * N_;
  for (int j = t16; j < N_; j += 16) arow[j] = (__bf16)(sc[row * N_ + j] * inv);
}

// ---------------------------------------------------------------------------
// O = A @ Vcat per (b,h), 4x i-tile reuse per wave (64 rows x 16 cols; Vcat
// fragment loaded once per j-step, feeds 4 WMMAs). Scatter into concat layout
// [B, N, 2944], replicating the reference's raw o_pair reshape:
//   (b,h,i,d) -> out[b, h*128 + i/8, 2816 + (i&7)*16 + d]
// ---------------------------------------------------------------------------
__global__ void attn_out_kernel(const __bf16* __restrict__ Aatt,
                                const __bf16* __restrict__ Vcat,
                                float* __restrict__ Ocat) {
  int lane = threadIdx.x & 31;
  int wave = threadIdx.x >> 5;
  int tile = blockIdx.x * (blockDim.x >> 5) + wave;
  const int tiles_n = VW_ >> 4;  // 23
  if (tile >= B_ * H_ * 16 * tiles_n) return;
  int nt = tile % tiles_n;
  int rest = tile / tiles_n;
  int ig = rest & 15;            // 64-row i-group
  int h = (rest >> 4) & 7;
  int b = rest >> 7;
  int i0 = ig << 6, d0 = nt << 4;

  const __bf16* Ab = Aatt + (size_t)(b * H_ + h) * N_ * N_;
  const __bf16* Vb = Vcat + (size_t)(b * H_ + h) * N_ * VW_;
  v8f acc[4] = {v8f{}, v8f{}, v8f{}, v8f{}};
  for (int j0 = 0; j0 < N_; j0 += 32) {
    v16bf bfr = load_b_bf16(Vb, VW_, j0, d0, lane);
#pragma unroll
    for (int mt = 0; mt < 4; ++mt) {
      v16bf a = load_a_bf16(Ab, N_, i0 + 16 * mt, j0, lane);
      acc[mt] = wmma_bf16(a, bfr, acc[mt]);
    }
  }
  int g = lane >> 4, nn = lane & 15;
  int d = d0 + nn;
#pragma unroll
  for (int mt = 0; mt < 4; ++mt) {
#pragma unroll
    for (int rr = 0; rr < 8; ++rr) {
      int irow = i0 + 16 * mt + rr + 8 * g;
      size_t row, col;
      if (d < CH_) {               // o_scalar: [b, i, h*128 + d]
        row = irow; col = h * CH_ + d;
      } else if (d < CH_ + SH_) {  // o_sh: [b, i, 1024 + h*224 + (d-128)]
        row = irow; col = H_ * CH_ + h * SH_ + (d - CH_);
      } else {                     // o_pair raw reshape scatter
        row = h * 128 + (irow >> 3);
        col = H_ * CH_ + H_ * SH_ + ((irow & 7) << 4) + (d - CH_ - SH_);
      }
      Ocat[((size_t)b * N_ + row) * OCAT_ + col] = acc[mt][rr];
    }
  }
}

// ---------------------------------------------------------------------------
extern "C" void kernel_launch(void* const* d_in, const int* in_sizes, int n_in,
                              void* d_out, int out_size, void* d_ws, size_t ws_size,
                              hipStream_t stream) {
  (void)in_sizes; (void)n_in; (void)out_size; (void)ws_size;
  const float* s    = (const float*)d_in[0];
  const float* z    = (const float*)d_in[1];
  const float* r    = (const float*)d_in[2];
  const float* mask = (const float*)d_in[3];
  const float* Wq   = (const float*)d_in[4];
  const float* bq   = (const float*)d_in[5];
  const float* Wkv  = (const float*)d_in[6];
  const float* bkv  = (const float*)d_in[7];
  const float* Wqsh = (const float*)d_in[8];
  const float* bqsh = (const float*)d_in[9];
  const float* Wkvsh= (const float*)d_in[10];
  const float* bkvsh= (const float*)d_in[11];
  const float* Wb   = (const float*)d_in[12];
  const float* bb   = (const float*)d_in[13];
  const float* Wdz  = (const float*)d_in[14];
  const float* bdz  = (const float*)d_in[15];
  const float* Wout = (const float*)d_in[16];
  const float* bout = (const float*)d_in[17];

  // workspace carve-out (~230 MB)
  char* ws = (char*)d_ws;
  size_t off = 0;
  auto carve = [&](size_t bytes) {
    char* p = ws + off;
    off = (off + bytes + 255) & ~(size_t)255;
    return p;
  };
  float*  q     = (float*) carve((size_t)B_ * N_ * H_ * CH_ * 4);        // 8 MB
  float*  kv    = (float*) carve((size_t)B_ * N_ * H_ * 2 * CH_ * 4);    // 16 MB
  float*  qsh   = (float*) carve((size_t)B_ * N_ * H_ * SH_ * 4);        // 14 MB
  float*  kvsh  = (float*) carve((size_t)B_ * N_ * 2 * H_ * SH_ * 4);    // 29 MB
  float*  bias2 = (float*) carve((size_t)B_ * H_ * N_ * N_ * 4);         // 64 MB
  float*  pz    = (float*) carve((size_t)B_ * N_ * DZ_ * 4);             // 128 KB
  __bf16* Qcat  = (__bf16*)carve((size_t)B_ * H_ * N_ * QKW_ * 2);       // 11.5 MB
  __bf16* Kcat  = (__bf16*)carve((size_t)B_ * H_ * N_ * QKW_ * 2);       // 11.5 MB
  __bf16* Vcat  = (__bf16*)carve((size_t)B_ * H_ * N_ * VW_ * 2);        // 12 MB
  __bf16* Aatt  = (__bf16*)carve((size_t)B_ * H_ * N_ * N_ * 2);         // 32 MB
  float*  Ocat  = (float*) carve((size_t)B_ * N_ * OCAT_ * 4);           // 24 MB

  auto gblk = [](long waves) { return (int)((waves + 7) / 8); };
  const int M = B_ * N_;        // 2048
  const int MG = M / 64;        // 32 row-groups per GEMM

  // 1) input projections (bf16 WMMA GEMMs, 4x M reuse per wave)
  gemm_bias_kernel<<<gblk((long)MG*(1024/16)), 256, 0, stream>>>(s, Wq,   bq,   q,    M, 1024, CS_);
  gemm_bias_kernel<<<gblk((long)MG*(2048/16)), 256, 0, stream>>>(s, Wkv,  bkv,  kv,   M, 2048, CS_);
  gemm_bias_kernel<<<gblk((long)MG*(1792/16)), 256, 0, stream>>>(s, Wqsh, bqsh, qsh,  M, 1792, CS_);
  gemm_bias_kernel<<<gblk((long)MG*(3584/16)), 256, 0, stream>>>(s, Wkvsh,bkvsh,kvsh, M, 3584, CS_);

  // 2) single double-buffered async pass over z: bias (+mask) and pz
  z_fused_kernel<<<B_ * N_, 256, 2 * ZCHUNK_ * ZROW_PAD_ * sizeof(float), stream>>>(
      z, mask, Wb, bb, Wdz, bdz, bias2, pz);

  // 3) pack rotated/scaled bf16 operand tensors
  {
    long tq = (long)B_ * H_ * N_ * QKW_;
    pack_q_kernel<<<(int)((tq + 255) / 256), 256, 0, stream>>>(q, qsh, r, Qcat);
    pack_k_kernel<<<(int)((tq + 255) / 256), 256, 0, stream>>>(kv, kvsh, r, Kcat);
    long tv = (long)B_ * H_ * N_ * VW_;
    pack_v_kernel<<<(int)((tv + 255) / 256), 256, 0, stream>>>(kv, kvsh, r, pz, Vcat);
  }

  // 4) attention: scores+softmax (64 KB dynamic LDS), then O = A @ Vcat
  attn_scores_kernel<<<B_ * H_ * (N_ / 16), 256, 16 * N_ * sizeof(float), stream>>>(
      Qcat, Kcat, bias2, Aatt);
  attn_out_kernel<<<gblk((long)B_ * H_ * 16 * (VW_ / 16)), 256, 0, stream>>>(
      Aatt, Vcat, Ocat);

  // 5) output projection -> d_out
  gemm_bias_kernel<<<gblk((long)MG*(512/16)), 256, 0, stream>>>(
      Ocat, Wout, bout, (float*)d_out, M, CS_, OCAT_);
}
